// ScaledDotProductAttention_43413529428726
// MI455X (gfx1250) — matrix operand order repaired and checked by reading, not compiled, not run on hardware
//
#include <hip/hip_runtime.h>
#include <hip/hip_bf16.h>

typedef __bf16 bf16_t;
typedef __attribute__((ext_vector_type(16))) __bf16 v16bf;
typedef __attribute__((ext_vector_type(8)))  float  v8f;

#define B_  32
#define LQ_ 2048
#define LK_ 2048
#define D_  256
#define INV_TEMP 0.0625f   // 1/16
#define TQ  16             // q rows per workgroup
#define NW  8              // waves per workgroup (256 threads, wave32)
#define VSTR 40            // padded LDS stride (bf16 elems) for V staging rows

// Dynamic LDS layout:
//  [0,      24K) : phase0/1: Q tile bf16 [16][256] (8K); phase3: V staging [8][32][VSTR] bf16 (20K)
//  [24K,   152K) : S scores f32 [16][2048]
//  [152K,  216K) : P probs bf16 [16][2048]
static constexpr unsigned SMEM_BYTES = 216u * 1024u;

union FragB  { v16bf v; bf16_t e[16]; uint4 q[2]; };
union Pack16 { bf16_t e[16]; uint4 q[2]; };
union Pack4  { bf16_t e[4];  uint2 u;   };
union Pack2  { bf16_t e[2];  unsigned u; };

__device__ __forceinline__ float wave_max(float v) {
#pragma unroll
  for (int o = 16; o > 0; o >>= 1) v = fmaxf(v, __shfl_xor(v, o, 32));
  return v;
}
__device__ __forceinline__ float wave_sum(float v) {
#pragma unroll
  for (int o = 16; o > 0; o >>= 1) v += __shfl_xor(v, o, 32);
  return v;
}
__device__ __forceinline__ v8f zero8() {
  v8f z;
#pragma unroll
  for (int i = 0; i < 8; ++i) z[i] = 0.0f;
  return z;
}
__device__ __forceinline__ v8f wmma_bf16(v16bf a, v16bf b, v8f c) {
  return __builtin_amdgcn_wmma_f32_16x16x32_bf16(false, a, false, b, (short)0, c,
                                                 false, false);
}

__global__ __launch_bounds__(256, 1)
void attn_fused_kernel(const float* __restrict__ Q, const float* __restrict__ K,
                       const float* __restrict__ V,
                       const unsigned char* __restrict__ src_mask,
                       const unsigned char* __restrict__ mel_mask,
                       float* __restrict__ out, float* __restrict__ attn) {
  extern __shared__ char smem[];
  bf16_t* ls_q = (bf16_t*)smem;                 // [16][256] bf16 (phase 0/1)
  bf16_t* ls_v = (bf16_t*)smem;                 // [NW][32][VSTR] bf16 (phase 3)
  float*  ls_s = (float*)(smem + 24 * 1024);    // [16][2048] f32
  bf16_t* ls_p = (bf16_t*)(smem + 152 * 1024);  // [16][2048] bf16

  const int tid = threadIdx.x;
  const int w   = tid >> 5;   // wave id 0..7
  const int l   = tid & 31;   // lane
  const int h   = l >> 4;     // lane half (WMMA layout)
  const int n   = l & 15;     // lane-within-half: B-col / A-row / C-col

  const int b     = blockIdx.x >> 7;          // 128 q-tiles per batch
  const int qbase = (blockIdx.x & 127) * TQ;

  // -------- Phase 0: Q tile (16x256 f32) -> LDS bf16 --------
  {
    const int row  = tid >> 4;
    const int col0 = (tid & 15) * 16;
    const float* qp = Q + (size_t)(b * LQ_ + qbase + row) * D_ + col0;
    Pack16 t;
#pragma unroll
    for (int j = 0; j < 4; ++j) {
      float4 f = *(const float4*)(qp + j * 4);
      t.e[j * 4 + 0] = (bf16_t)f.x; t.e[j * 4 + 1] = (bf16_t)f.y;
      t.e[j * 4 + 2] = (bf16_t)f.z; t.e[j * 4 + 3] = (bf16_t)f.w;
    }
    *(uint4*)&ls_q[row * D_ + col0]     = t.q[0];
    *(uint4*)&ls_q[row * D_ + col0 + 8] = t.q[1];
  }
  __syncthreads();

  // -------- Phase 1: S = Q K^T via v_wmma_f32_16x16x32_bf16 --------
  // Wave w owns S columns [w*256, w*256+256): 16 tiles of 16x16, K-dim = 256 -> 8 WMMA steps.
  {
    FragB afrag[8];  // Q fragments reused across all 16 tiles (A 16x32 bf16, contiguous k = 16h + s)
#pragma unroll
    for (int d = 0; d < 8; ++d) {
      const int dstep = d * 32;
      afrag[d].q[0] = *(const uint4*)&ls_q[n * D_ + dstep + 16 * h];
      afrag[d].q[1] = *(const uint4*)&ls_q[n * D_ + dstep + 16 * h + 8];
    }
    for (int t = 0; t < 16; ++t) {
      const int kcol = w * 256 + t * 16;
      // B fragment lane l holds K-matrix row (kcol+n), d-range 16h..16h+15 per step
      const float* kp = K + (size_t)(b * LK_ + kcol + n) * D_;
      if (t < 15) __builtin_prefetch(kp + 16 * D_, 0, 0);  // next tile's K row
      v8f acc = zero8();
#pragma unroll
      for (int d = 0; d < 8; ++d) {
        const int dstep = d * 32 + 16 * h;
        FragB bf;
#pragma unroll
        for (int j = 0; j < 4; ++j) {
          float4 f = *(const float4*)(kp + dstep + j * 4);
          bf.e[j * 4 + 0] = (bf16_t)f.x; bf.e[j * 4 + 1] = (bf16_t)f.y;
          bf.e[j * 4 + 2] = (bf16_t)f.z; bf.e[j * 4 + 3] = (bf16_t)f.w;
        }
        acc = wmma_bf16(afrag[d].v, bf.v, acc);
      }
#pragma unroll
      for (int r = 0; r < 8; ++r)        // C/D layout: VGPR r -> row r+8h, col n
        ls_s[(r + 8 * h) * LK_ + kcol + n] = acc[r];
    }
  }
  __syncthreads();

  // -------- Phase 2: softmax with dual masks; write attn; P -> LDS bf16 --------
  {
    const float NEG = -__builtin_inff();
    for (int rr = 0; rr < 2; ++rr) {      // wave w handles rows 2w, 2w+1
      const int q = w * 2 + rr;
      const size_t rowoff = (size_t)(b * LQ_ + qbase + q) * (size_t)LK_;
      const unsigned char* melr = mel_mask + rowoff;
      const unsigned char* srcr = src_mask + rowoff;
      float* attnr = attn + rowoff;
      float* srow  = ls_s + q * LK_;

      float rmax = NEG;
#pragma unroll
      for (int i = 0; i < 16; ++i) {
        const int c0 = i * 128 + l * 4;
        uchar4 mm = *(const uchar4*)(melr + c0);
        float4 sv = *(float4*)&srow[c0];
        sv.x = mm.x ? NEG : sv.x * INV_TEMP;
        sv.y = mm.y ? NEG : sv.y * INV_TEMP;
        sv.z = mm.z ? NEG : sv.z * INV_TEMP;
        sv.w = mm.w ? NEG : sv.w * INV_TEMP;
        rmax = fmaxf(rmax, fmaxf(fmaxf(sv.x, sv.y), fmaxf(sv.z, sv.w)));
        *(float4*)&srow[c0] = sv;         // keep scaled+masked scores
      }
      rmax = wave_max(rmax);
      if (!(rmax > -1e30f)) rmax = 0.0f;  // guard fully-masked row

      float rsum = 0.0f;
#pragma unroll
      for (int i = 0; i < 16; ++i) {
        const int c0 = i * 128 + l * 4;
        float4 sv = *(float4*)&srow[c0];
        rsum += __expf(sv.x - rmax) + __expf(sv.y - rmax) +
                __expf(sv.z - rmax) + __expf(sv.w - rmax);
      }
      rsum = wave_sum(rsum);
      const float inv = 1.0f / fmaxf(rsum, 1e-30f);

#pragma unroll
      for (int i = 0; i < 16; ++i) {
        const int c0 = i * 128 + l * 4;
        float4 sv = *(float4*)&srow[c0];
        uchar4 ss = *(const uchar4*)(srcr + c0);
        float4 p;
        p.x = ss.x ? 0.0f : __expf(sv.x - rmax) * inv;
        p.y = ss.y ? 0.0f : __expf(sv.y - rmax) * inv;
        p.z = ss.z ? 0.0f : __expf(sv.z - rmax) * inv;
        p.w = ss.w ? 0.0f : __expf(sv.w - rmax) * inv;
        *(float4*)(attnr + c0) = p;       // materialize attn (src-masked)
        Pack4 pk;
        pk.e[0] = (bf16_t)p.x; pk.e[1] = (bf16_t)p.y;
        pk.e[2] = (bf16_t)p.z; pk.e[3] = (bf16_t)p.w;
        *(uint2*)&ls_p[q * LK_ + c0] = pk.u;
      }
    }
  }
  __syncthreads();

  // -------- Phase 3: out = P @ V via WMMA; wave w owns output cols [32w,32w+32) --------
  // V staged TRANSPOSED in LDS: lvw[d_local][k_local], row stride VSTR (padded).
  // Each lane loads two adjacent V rows (k=2n, 2n+1) so each (d) pair packs into one
  // b32 LDS write; B fragments then read as two contiguous ds_read_b128.
  {
    bf16_t* myv = ls_v + w * (32 * VSTR);   // per-wave private staging
    v8f acc0 = zero8(), acc1 = zero8();
    for (int kk = 0; kk < LK_; kk += 32) {
      // lane l: V rows kk+2n, kk+2n+1, d-range 32w + 16h .. +15
      const float* vp0 = V + (size_t)(b * LK_ + kk + 2 * n) * D_ + w * 32 + 16 * h;
      const float* vp1 = vp0 + D_;
      if (kk + 32 < LK_) __builtin_prefetch(vp0 + 32 * D_, 0, 0);  // next chunk
      float4 r0a = *(const float4*)(vp0);
      float4 r0b = *(const float4*)(vp0 + 4);
      float4 r0c = *(const float4*)(vp0 + 8);
      float4 r0d = *(const float4*)(vp0 + 12);
      float4 r1a = *(const float4*)(vp1);
      float4 r1b = *(const float4*)(vp1 + 4);
      float4 r1c = *(const float4*)(vp1 + 8);
      float4 r1d = *(const float4*)(vp1 + 12);
      float e0[16] = {r0a.x, r0a.y, r0a.z, r0a.w, r0b.x, r0b.y, r0b.z, r0b.w,
                      r0c.x, r0c.y, r0c.z, r0c.w, r0d.x, r0d.y, r0d.z, r0d.w};
      float e1[16] = {r1a.x, r1a.y, r1a.z, r1a.w, r1b.x, r1b.y, r1b.z, r1b.w,
                      r1c.x, r1c.y, r1c.z, r1c.w, r1d.x, r1d.y, r1d.z, r1d.w};
#pragma unroll
      for (int d = 0; d < 16; ++d) {      // d_local = 16h + d ; k_local = 2n, 2n+1
        Pack2 pk;
        pk.e[0] = (bf16_t)e0[d];          // lo = k even  (v_cvt_pk_bf16_f32)
        pk.e[1] = (bf16_t)e1[d];          // hi = k odd
        *(unsigned*)&myv[(16 * h + d) * VSTR + 2 * n] = pk.u;
      }

      // A fragment from P (row m=n, K-dim = k index)
      FragB af;
      af.q[0] = *(const uint4*)&ls_p[n * LK_ + kk + 16 * h];
      af.q[1] = *(const uint4*)&ls_p[n * LK_ + kk + 16 * h + 8];

      // B fragments: tile c column d_local=16c+n, K-range 16h..16h+15 -> contiguous
      FragB b0, b1;
      b0.q[0] = *(const uint4*)&myv[(n)*VSTR + 16 * h];
      b0.q[1] = *(const uint4*)&myv[(n)*VSTR + 16 * h + 8];
      b1.q[0] = *(const uint4*)&myv[(16 + n) * VSTR + 16 * h];
      b1.q[1] = *(const uint4*)&myv[(16 + n) * VSTR + 16 * h + 8];

      acc0 = wmma_bf16(af.v, b0.v, acc0);
      acc1 = wmma_bf16(af.v, b1.v, acc1);
    }
#pragma unroll
    for (int r = 0; r < 8; ++r) {
      const size_t o = (size_t)(b * LQ_ + qbase + r + 8 * h) * D_ + w * 32 + n;
      out[o]      = acc0[r];
      out[o + 16] = acc1[r];
    }
  }
}

extern "C" void kernel_launch(void* const* d_in, const int* in_sizes, int n_in,
                              void* d_out, int out_size, void* d_ws, size_t ws_size,
                              hipStream_t stream) {
  (void)in_sizes; (void)n_in; (void)d_ws; (void)ws_size; (void)out_size;
  const float* q = (const float*)d_in[0];
  const float* k = (const float*)d_in[1];
  const float* v = (const float*)d_in[2];
  const unsigned char* src_mask = (const unsigned char*)d_in[3];
  const unsigned char* mel_mask = (const unsigned char*)d_in[4];
  float* out  = (float*)d_out;
  float* attn = out + (size_t)B_ * LQ_ * D_;   // outputs concatenated: (output, attn)

  hipFuncSetAttribute((const void*)attn_fused_kernel,
                      hipFuncAttributeMaxDynamicSharedMemorySize, (int)SMEM_BYTES);

  dim3 grid(B_ * (LQ_ / TQ));  // 4096 workgroups
  dim3 block(256);             // 8 wave32s
  attn_fused_kernel<<<grid, block, SMEM_BYTES, stream>>>(q, k, v, src_mask,
                                                         mel_mask, out, attn);
}
